// TacotronDecoder_28750511079668
// MI455X (gfx1250) — matrix operand order repaired and checked by reading, not compiled, not run on hardware
//
#include <hip/hip_runtime.h>
#include <hip/hip_bf16.h>

// ---------------------------------------------------------------------------
// Tacotron decoder for MI455X (gfx1250):
//   Phase 1 (parallel): bf16-pack weights (per-WG gate-column slices), prenet
//                       for all 400 steps (teacher forcing!), pack x_all.
//   Phase 2 (serial)  : persistent 128-WG kernel, 400 LSTM steps, weights
//                       LDS-resident (240KB/WG), bf16 WMMA 16x16x32, h state
//                       exchanged through L2 with global monotonic barriers.
//   Phase 3 (parallel): feat_out projection as one 25600x160x1536 WMMA GEMM,
//                       att_w filled with hash-based uniforms (ref uses rand).
// ---------------------------------------------------------------------------

typedef __attribute__((ext_vector_type(16))) __bf16 v16bf;
typedef __attribute__((ext_vector_type(8)))  float  v8f;
typedef unsigned int u32x4 __attribute__((ext_vector_type(4)));

union Frag { v16bf v; u32x4 q[2]; };

#define NWG     128
#define TSTEPS  400
#define WSLICE  122880          // ushorts per WG: 32*1792 (cell0) + 32*2048 (cell1)
#define C0K     1792
#define C1K     2048

__device__ __forceinline__ unsigned short f2bf(float f) {
    unsigned u = __float_as_uint(f);
    unsigned r = u + 0x7FFFu + ((u >> 16) & 1u);   // RNE
    return (unsigned short)(r >> 16);
}
__device__ __forceinline__ float bf2f(unsigned short u) {
    return __uint_as_float(((unsigned)u) << 16);
}
__device__ __forceinline__ float sigf(float x) { return 1.0f / (1.0f + expf(-x)); }

// ------------------------------- utility -----------------------------------
__global__ void k_zero(unsigned* __restrict__ p, int n) {
    int i = blockIdx.x * 256 + threadIdx.x;
    if (i < n) p[i] = 0u;
}

// --------------------- weight / bias / input packing -----------------------
// wpack layout: [wg][cell0: 32 cols x 1792][cell1: 32 cols x 2048], bf16.
// col c (0..31) of a cell -> gate g=c>>3, hidden unit j = wg*8 + (c&7),
// global gate column = g*1024 + j (torch i,f,g,o order).
__global__ void k_pack_w(const float* __restrict__ wih0, const float* __restrict__ whh0,
                         const float* __restrict__ wih1, const float* __restrict__ whh1,
                         unsigned short* __restrict__ wpack) {
    size_t e = (size_t)blockIdx.x * 256 + threadIdx.x;
    if (e >= (size_t)NWG * WSLICE) return;
    int wg = (int)(e / WSLICE);
    int r  = (int)(e % WSLICE);
    float v;
    if (r < 32 * C0K) {
        int c = r / C0K, k = r % C0K;
        int col = (c >> 3) * 1024 + wg * 8 + (c & 7);
        v = (k < 768) ? wih0[(size_t)col * 768 + k] : whh0[(size_t)col * 1024 + (k - 768)];
    } else {
        int r2 = r - 32 * C0K;
        int c = r2 / C1K, k = r2 % C1K;
        int col = (c >> 3) * 1024 + wg * 8 + (c & 7);
        v = (k < 1024) ? wih1[(size_t)col * 1024 + k] : whh1[(size_t)col * 1024 + (k - 1024)];
    }
    wpack[e] = f2bf(v);
}

__global__ void k_pack_bias(const float* __restrict__ bih0, const float* __restrict__ bhh0,
                            const float* __restrict__ bih1, const float* __restrict__ bhh1,
                            float* __restrict__ bias) {
    int e = blockIdx.x * 256 + threadIdx.x;
    if (e >= NWG * 64) return;
    int wg = e >> 6, c = e & 63;
    int cell = c >> 5, cc = c & 31;
    int col = (cc >> 3) * 1024 + wg * 8 + (cc & 7);
    bias[e] = cell ? (bih1[col] + bhh1[col]) : (bih0[col] + bhh0[col]);
}

__global__ void k_pack_fw(const float* __restrict__ fow, unsigned short* __restrict__ fwb) {
    int i = blockIdx.x * 256 + threadIdx.x;
    if (i < 160 * 1536) fwb[i] = f2bf(fow[i]);
}

// prenet layer 1: rows = t*64+b (25600), 256 units; prev_out = tgt[t-1] (0 at t=0)
__global__ void k_prenet1(const float* __restrict__ ft, const float* __restrict__ w1,
                          const float* __restrict__ b1, float* __restrict__ ph1) {
    int row = blockIdx.x;                 // t*64 + b
    int t = row >> 6, b = row & 63;
    int u = threadIdx.x;                  // 0..255
    float s = b1[u];
    if (t > 0) {
        // prev_out[j] = feature_target[b, j%80, (t-1)*2 + j/80]
        const float* fb = ft + (size_t)b * 64000 + (size_t)(t - 1) * 2;
        const float* wr = w1 + (size_t)u * 160;
        for (int o = 0; o < 80; ++o) s += wr[o]      * fb[(size_t)o * 800];
        for (int o = 0; o < 80; ++o) s += wr[80 + o] * fb[(size_t)o * 800 + 1];
    }
    ph1[(size_t)row * 256 + u] = fmaxf(s, 0.0f);
}

// prenet layer 2 -> x_all[row][512 + u] (bf16)
__global__ void k_prenet2(const float* __restrict__ ph1, const float* __restrict__ w2,
                          const float* __restrict__ b2, unsigned short* __restrict__ xall) {
    int row = blockIdx.x;
    int u = threadIdx.x;
    const float* pr = ph1 + (size_t)row * 256;
    const float* wr = w2 + (size_t)u * 256;
    float s = b2[u];
    for (int k = 0; k < 256; ++k) s += wr[k] * pr[k];
    xall[(size_t)row * 768 + 512 + u] = f2bf(fmaxf(s, 0.0f));
}

// enc_output (B,T,512) -> x_all[row=t*64+b][0..511] (bf16)
__global__ void k_packenc(const float* __restrict__ enc, unsigned short* __restrict__ xall) {
    size_t i = (size_t)blockIdx.x * 256 + threadIdx.x;
    if (i >= (size_t)25600 * 512) return;
    int row = (int)(i >> 9), k = (int)(i & 511);
    int t = row >> 6, b = row & 63;
    xall[(size_t)row * 768 + k] = f2bf(enc[(size_t)b * 204800 + (size_t)t * 512 + k]);
}

// ------------------------- persistent LSTM kernel --------------------------
__device__ __forceinline__ void grid_sync(unsigned* __restrict__ bar, unsigned* phase, int tid) {
    __syncthreads();
    if (tid == 0) {
        unsigned target = *phase + 1u;
        __builtin_amdgcn_fence(__ATOMIC_RELEASE, "agent");
        unsigned old = __hip_atomic_fetch_add(&bar[0], 1u, __ATOMIC_RELAXED,
                                              __HIP_MEMORY_SCOPE_AGENT);
        if (old == target * NWG - 1u) {
            __hip_atomic_store(&bar[1], target, __ATOMIC_RELEASE, __HIP_MEMORY_SCOPE_AGENT);
        } else {
            while (__hip_atomic_load(&bar[1], __ATOMIC_ACQUIRE, __HIP_MEMORY_SCOPE_AGENT) < target)
                __builtin_amdgcn_s_sleep(1);
        }
        __builtin_amdgcn_fence(__ATOMIC_ACQUIRE, "agent");
        *phase = target;
    }
    __syncthreads();
}

// One wave per 16x16 WMMA output tile: wave w -> mtile=w&3 (batch), nt=w>>2 (16 cols).
// A frag: lane m=lane&15, half=lane>>4; two 16B runs at k0+8*half and k0+16+8*half.
// B frag: lane n=lane&15, half=lane>>4; two 16B runs at k0+8*half and k0+16+8*half of row n of LDS weight slice.
__global__ void __launch_bounds__(256)
k_decode(const unsigned short* __restrict__ wpack, const float* __restrict__ bias,
         const unsigned short* __restrict__ xall,
         unsigned short* __restrict__ hb0, unsigned short* __restrict__ hb1,
         unsigned short* __restrict__ h1all, unsigned* __restrict__ bar) {
    extern __shared__ unsigned short lds_w[];        // 245,760 B: persistent weights
    __shared__ float gates[64 * 32];
    __shared__ float c0s[64 * 8];
    __shared__ float c1s[64 * 8];

    const int tid = threadIdx.x;
    const int wg  = blockIdx.x;

    // Preload this WG's weight slice into LDS once (stays for all 400 steps).
    {
        const u32x4* src = (const u32x4*)(wpack + (size_t)wg * WSLICE);
        u32x4* dst = (u32x4*)lds_w;
        for (int i = tid; i < WSLICE / 8; i += 256) dst[i] = src[i];
    }
    for (int i = tid; i < 512; i += 256) { c0s[i] = 0.0f; c1s[i] = 0.0f; }
    __syncthreads();

    const int lane  = tid & 31;
    const int wv    = tid >> 5;
    const int mtile = wv & 3;
    const int nt    = wv >> 2;
    const int ml    = lane & 15;
    const int half  = lane >> 4;
    const int nn    = lane & 15;
    const unsigned short* ldsw0 = lds_w;
    const unsigned short* ldsw1 = lds_w + 32 * C0K;
    unsigned phase = 0;

    for (int t = 0; t < TSTEPS; ++t) {
        const int cur = t & 1, nxt = cur ^ 1;
        const unsigned short* xbase = xall + (size_t)t * (64 * 768);
        const unsigned short* h0c = hb0 + (size_t)cur * 65536;
        unsigned short*       h0n = hb0 + (size_t)nxt * 65536;
        const unsigned short* h1c = hb1 + (size_t)cur * 65536;
        unsigned short*       h1n = hb1 + (size_t)nxt * 65536;

        if (t + 1 < TSTEPS)   // warm L0/L2 for next step's activations
            __builtin_prefetch(xall + (size_t)(t + 1) * (64 * 768) +
                               (size_t)(mtile * 16 + ml) * 768, 0, 1);

        // ---------------- cell 0: K = 768 (x_t) + 1024 (h0_prev) ----------------
        {
            v8f acc = {0.f, 0.f, 0.f, 0.f, 0.f, 0.f, 0.f, 0.f};
            const unsigned short* ar0  = xbase + (size_t)(mtile * 16 + ml) * 768 + 8 * half;
            const unsigned short* ar1  = h0c   + (size_t)(mtile * 16 + ml) * 1024 + 8 * half;
            const unsigned short* brow = ldsw0 + (size_t)(nt * 16 + nn) * C0K + 8 * half;
            for (int kt = 0; kt < 56; ++kt) {
                const unsigned short* ap = (kt < 24) ? (ar0 + kt * 32) : (ar1 + (kt - 24) * 32);
                Frag a, b;
                a.q[0] = *(const u32x4*)ap;
                a.q[1] = *(const u32x4*)(ap + 16);
                const unsigned short* bp = brow + kt * 32;
                b.q[0] = *(const u32x4*)bp;
                b.q[1] = *(const u32x4*)(bp + 16);
                acc = __builtin_amdgcn_wmma_f32_16x16x32_bf16(false, a.v, false, b.v,
                                                              (short)0, acc, false, false);
            }
#pragma unroll
            for (int r = 0; r < 8; ++r)
                gates[(mtile * 16 + 8 * half + r) * 32 + nt * 16 + nn] = acc[r];
        }
        __syncthreads();
        for (int it = tid; it < 512; it += 256) {        // pointwise + zoneout, cell0
            const int m = it & 63, j = it >> 6, unit = wg * 8 + j;
            float gi = gates[m * 32 + j]       + bias[wg * 64 + j];
            float gf = gates[m * 32 + 8 + j]   + bias[wg * 64 + 8 + j];
            float gg = gates[m * 32 + 16 + j]  + bias[wg * 64 + 16 + j];
            float go = gates[m * 32 + 24 + j]  + bias[wg * 64 + 24 + j];
            float cp = c0s[m * 8 + j];
            float cn = sigf(gf) * cp + sigf(gi) * tanhf(gg);
            float hn = sigf(go) * tanhf(cn);
            float hp = bf2f(h0c[(size_t)m * 1024 + unit]);
            cn = 0.1f * cp + 0.9f * cn;                  // zoneout (eval mode)
            hn = 0.1f * hp + 0.9f * hn;
            c0s[m * 8 + j] = cn;
            h0n[(size_t)m * 1024 + unit] = f2bf(hn);
        }
        __syncthreads();
        grid_sync(bar, &phase, tid);                     // h0_new visible everywhere

        // ---------------- cell 1: K = 1024 (h0_new) + 1024 (h1_prev) ------------
        {
            v8f acc = {0.f, 0.f, 0.f, 0.f, 0.f, 0.f, 0.f, 0.f};
            const unsigned short* ar0  = h0n + (size_t)(mtile * 16 + ml) * 1024 + 8 * half;
            const unsigned short* ar1  = h1c + (size_t)(mtile * 16 + ml) * 1024 + 8 * half;
            const unsigned short* brow = ldsw1 + (size_t)(nt * 16 + nn) * C1K + 8 * half;
            for (int kt = 0; kt < 64; ++kt) {
                const unsigned short* ap = (kt < 32) ? (ar0 + kt * 32) : (ar1 + (kt - 32) * 32);
                Frag a, b;
                a.q[0] = *(const u32x4*)ap;
                a.q[1] = *(const u32x4*)(ap + 16);
                const unsigned short* bp = brow + kt * 32;
                b.q[0] = *(const u32x4*)bp;
                b.q[1] = *(const u32x4*)(bp + 16);
                acc = __builtin_amdgcn_wmma_f32_16x16x32_bf16(false, a.v, false, b.v,
                                                              (short)0, acc, false, false);
            }
#pragma unroll
            for (int r = 0; r < 8; ++r)
                gates[(mtile * 16 + 8 * half + r) * 32 + nt * 16 + nn] = acc[r];
        }
        __syncthreads();
        for (int it = tid; it < 512; it += 256) {        // pointwise + zoneout, cell1
            const int m = it & 63, j = it >> 6, unit = wg * 8 + j;
            float gi = gates[m * 32 + j]       + bias[wg * 64 + 32 + j];
            float gf = gates[m * 32 + 8 + j]   + bias[wg * 64 + 40 + j];
            float gg = gates[m * 32 + 16 + j]  + bias[wg * 64 + 48 + j];
            float go = gates[m * 32 + 24 + j]  + bias[wg * 64 + 56 + j];
            float cp = c1s[m * 8 + j];
            float cn = sigf(gf) * cp + sigf(gi) * tanhf(gg);
            float hn = sigf(go) * tanhf(cn);
            float hp = bf2f(h1c[(size_t)m * 1024 + unit]);
            cn = 0.1f * cp + 0.9f * cn;
            hn = 0.1f * hp + 0.9f * hn;
            c1s[m * 8 + j] = cn;
            unsigned short hb = f2bf(hn);
            h1n[(size_t)m * 1024 + unit] = hb;
            h1all[((size_t)t * 64 + m) * 1024 + unit] = hb;   // for epilogue GEMM
        }
        __syncthreads();
        grid_sync(bar, &phase, tid);                     // h1_new visible everywhere
    }
}

// -------------------- epilogue: feat_out projection (WMMA) -----------------
// out (25600 x 160) = [h1_all | enc] (25600 x 1536) @ fwb^T, scattered into
// output[b, o, t*2 + r] with j = r*80 + o.
__global__ void __launch_bounds__(256)
k_featout(const unsigned short* __restrict__ h1all, const unsigned short* __restrict__ xall,
          const unsigned short* __restrict__ fwb, float* __restrict__ out) {
    const int tid = threadIdx.x, lane = tid & 31, wv = tid >> 5;
    const int tileid = blockIdx.x * 8 + wv;        // 1600 Mtiles * 10 Ntiles
    const int mtile = tileid / 10, ntile = tileid % 10;
    const int ml = lane & 15, half = lane >> 4, nn = lane & 15;
    const int mrow = mtile * 16 + ml;
    const unsigned short* ha   = h1all + (size_t)mrow * 1024 + 8 * half;
    const unsigned short* xa   = xall  + (size_t)mrow * 768  + 8 * half;   // enc part (cols 0..511)
    const unsigned short* brow = fwb + (size_t)(ntile * 16 + nn) * 1536 + 8 * half;
    v8f acc = {0.f, 0.f, 0.f, 0.f, 0.f, 0.f, 0.f, 0.f};
    for (int kt = 0; kt < 48; ++kt) {              // 32 tiles h1 (K<1024) + 16 tiles enc
        const unsigned short* ap = (kt < 32) ? (ha + kt * 32) : (xa + (kt - 32) * 32);
        Frag a, b;
        a.q[0] = *(const u32x4*)ap;
        a.q[1] = *(const u32x4*)(ap + 16);
        const unsigned short* bp = brow + kt * 32;
        b.q[0] = *(const u32x4*)bp;
        b.q[1] = *(const u32x4*)(bp + 16);
        acc = __builtin_amdgcn_wmma_f32_16x16x32_bf16(false, a.v, false, b.v,
                                                      (short)0, acc, false, false);
    }
    const int j = ntile * 16 + nn;
    const int rfr = j / 80, o = j % 80;
#pragma unroll
    for (int r = 0; r < 8; ++r) {
        int m = mtile * 16 + 8 * half + r;
        int t = m >> 6, b = m & 63;
        out[(size_t)b * 64000 + (size_t)o * 800 + (size_t)t * 2 + rfr] = acc[r];
    }
}

// att_w: reference emits fresh uniform randoms (no fixed seed semantics we can
// reproduce bit-exactly); fill with a deterministic hash-based uniform in [0,1).
__global__ void k_attw(float* __restrict__ aw, int n) {
    int i = blockIdx.x * 256 + threadIdx.x;
    if (i >= n) return;
    unsigned s = (unsigned)i * 747796405u + 2891336453u;
    s = ((s >> ((s >> 28) + 4u)) ^ s) * 277803737u;
    s = (s >> 22) ^ s;
    aw[i] = (float)(s >> 8) * (1.0f / 16777216.0f);
}

// ------------------------------- host side ---------------------------------
extern "C" void kernel_launch(void* const* d_in, const int* in_sizes, int n_in,
                              void* d_out, int out_size, void* d_ws, size_t ws_size,
                              hipStream_t stream) {
    const float* enc  = (const float*)d_in[0];
    const float* ft   = (const float*)d_in[1];
    // d_in[2] lip_len: unused by the reference forward
    const float* pw1  = (const float*)d_in[3];
    const float* pb1  = (const float*)d_in[4];
    const float* pw2  = (const float*)d_in[5];
    const float* pb2  = (const float*)d_in[6];
    const float* wih0 = (const float*)d_in[7];
    const float* whh0 = (const float*)d_in[8];
    const float* bih0 = (const float*)d_in[9];
    const float* bhh0 = (const float*)d_in[10];
    const float* wih1 = (const float*)d_in[11];
    const float* whh1 = (const float*)d_in[12];
    const float* bih1 = (const float*)d_in[13];
    const float* bhh1 = (const float*)d_in[14];
    const float* fow  = (const float*)d_in[15];

    char* base = (char*)d_ws;
    size_t off = 0;
    auto carve = [&](size_t bytes) -> void* {
        void* p = base + off;
        off = (off + bytes + 255) & ~(size_t)255;
        return p;
    };
    unsigned*       bar   = (unsigned*)      carve(256);
    unsigned short* wpack = (unsigned short*)carve((size_t)NWG * WSLICE * 2);
    float*          bias  = (float*)         carve((size_t)NWG * 64 * 4);
    unsigned short* fwb   = (unsigned short*)carve((size_t)160 * 1536 * 2);
    unsigned short* xall  = (unsigned short*)carve((size_t)25600 * 768 * 2);
    unsigned short* h1all = (unsigned short*)carve((size_t)25600 * 1024 * 2);
    unsigned short* hb0   = (unsigned short*)carve((size_t)2 * 65536 * 2);
    unsigned short* hb1   = (unsigned short*)carve((size_t)2 * 65536 * 2);
    float*          ph1   = (float*)         carve((size_t)25600 * 256 * 4);
    (void)ws_size; (void)in_sizes; (void)n_in; (void)out_size;

    float* out = (float*)d_out;                  // [0, 4,096,000): output
    float* aw  = out + 4096000;                  // [.., +6,400,000): att_w

    // Phase 0: reset barrier + h state (workspace is not re-zeroed by harness).
    k_zero<<<1, 256, 0, stream>>>(bar, 64);
    k_zero<<<256, 256, 0, stream>>>((unsigned*)hb0, 65536);
    k_zero<<<256, 256, 0, stream>>>((unsigned*)hb1, 65536);

    // Phase 1: packing + prenet (all parallel; teacher forcing removes the
    // prenet from the serial loop).
    {
        size_t n = (size_t)NWG * WSLICE;
        k_pack_w<<<(unsigned)((n + 255) / 256), 256, 0, stream>>>(wih0, whh0, wih1, whh1, wpack);
    }
    k_pack_bias<<<(NWG * 64 + 255) / 256, 256, 0, stream>>>(bih0, bhh0, bih1, bhh1, bias);
    k_pack_fw<<<(160 * 1536 + 255) / 256, 256, 0, stream>>>(fow, fwb);
    k_prenet1<<<25600, 256, 0, stream>>>(ft, pw1, pb1, ph1);
    k_prenet2<<<25600, 256, 0, stream>>>(ph1, pw2, pb2, xall);
    {
        size_t n = (size_t)25600 * 512;
        k_packenc<<<(unsigned)((n + 255) / 256), 256, 0, stream>>>(enc, xall);
    }

    // Phase 2: persistent serial recurrence. 240KB dynamic LDS per WG holds
    // this WG's weight slices for all 400 steps.
    hipFuncSetAttribute((const void*)k_decode,
                        hipFuncAttributeMaxDynamicSharedMemorySize, WSLICE * 2);
    k_decode<<<NWG, 256, WSLICE * 2, stream>>>(wpack, bias, xall, hb0, hb1, h1all, bar);

    // Phase 3: output projection + att_w fill.
    k_featout<<<2000, 256, 0, stream>>>(h1all, xall, fwb, out);
    k_attw<<<(6400000 + 255) / 256, 256, 0, stream>>>(aw, 6400000);
}
